// GraphConv_38345468019333
// MI455X (gfx1250) — matrix operand order repaired and checked by reading, not compiled, not run on hardware
//
#include <hip/hip_runtime.h>

// Problem constants (from reference): B=4, N=50000, D=32, K_NBR=16, F=32
#define BB    4
#define NN    50000
#define DD    32
#define KNBR  16
#define FF    32
#define MT    32                      // rows per wave: two 16-row WMMA subtiles
#define NT32  ((NN + MT - 1) / MT)    // 1563 tiles (last tile: subtile1 invalid)
#define WELEM (KNBR * 2 * 32 * 16)    // 16384 bf16 elements per weight array

typedef __attribute__((ext_vector_type(16))) __bf16 v16bf;
typedef __attribute__((ext_vector_type(8)))  __bf16 v8bf;
typedef __attribute__((ext_vector_type(8)))  float  v8f;
typedef __attribute__((address_space(3))) unsigned short lds_us;

static __device__ __forceinline__ unsigned short f32_to_bf16_rn(float f) {
  unsigned int u = __float_as_uint(f);
  unsigned int r = u + 0x7FFFu + ((u >> 16) & 1u);   // round-to-nearest-even
  return (unsigned short)(r >> 16);
}
static __device__ __forceinline__ float bf16_bits_to_f32(unsigned short h) {
  return __uint_as_float(((unsigned int)h) << 16);
}

// ---- Prep 1: split x (fp32) into bf16 hi + lo arrays --------------------
__global__ void cvt_x_kernel(const float* __restrict__ x,
                             unsigned short* __restrict__ xhi,
                             unsigned short* __restrict__ xlo, int n) {
  int i = blockIdx.x * blockDim.x + threadIdx.x;
  if (i >= n) return;
  float f = x[i];
  unsigned short h = f32_to_bf16_rn(f);
  xhi[i] = h;
  xlo[i] = f32_to_bf16_rn(f - bf16_bits_to_f32(h));
}

// ---- Prep 2: pre-swizzle kernel weights into WMMA B-operand lane order --
// Layout: [k][f_half][lane][j] bf16, j=0..15 packed per lane.
// B (K=32 x N=16): lanes 0-15 hold K=0..7,16..23; lanes 16-31 hold K=8..15,24..31.
__global__ void pack_w_kernel(const float* __restrict__ w,
                              unsigned short* __restrict__ whi,
                              unsigned short* __restrict__ wlo) {
  int i = blockIdx.x * blockDim.x + threadIdx.x;   // 16384
  if (i >= WELEM) return;
  int j    = i & 15;
  int lane = (i >> 4) & 31;
  int fh   = (i >> 9) & 1;
  int k    = i >> 10;
  int col  = lane & 15;
  int K    = (j & 7) + 8 * (lane >> 4) + 16 * (j >> 3);
  float f = w[(k * DD + K) * FF + fh * 16 + col];  // kernel[k][d][f]
  unsigned short h = f32_to_bf16_rn(f);
  whi[i] = h;
  wlo[i] = f32_to_bf16_rn(f - bf16_bits_to_f32(h));
}

// ---- Main: gather + 16x16x32 bf16 WMMA, M=32/wave, weights in LDS -------
__global__ __launch_bounds__(256) void gconv_wmma_kernel(
    const unsigned short* __restrict__ xhi, const unsigned short* __restrict__ xlo,
    const unsigned short* __restrict__ whi, const unsigned short* __restrict__ wlo,
    const int* __restrict__ nbr2,   // int64 neighbor ids viewed as int pairs
    const float* __restrict__ bias, float* __restrict__ out) {
  __shared__ unsigned short sw[2 * WELEM];   // 64 KB: [0..]=hi, [WELEM..]=lo

  int t    = threadIdx.x;
  int lane = t & 31;

  // --- Stage weights into LDS via async memory->LDS DMA (ASYNCcnt path).
  // Every thread copies 2x8 chunks of 16B; barrier before use. Must run
  // before the tile guard so all waves reach the barrier.
  {
    unsigned lds_base = (unsigned)(unsigned long long)(lds_us*)sw;
#pragma unroll
    for (int half = 0; half < 2; ++half) {
      const char* src   = (const char*)(half ? wlo : whi);
      unsigned ldsoff   = lds_base + half * (WELEM * 2);
#pragma unroll
      for (int i = 0; i < 8; ++i) {
        unsigned boff = (unsigned)(t + i * 256) * 16u;
        unsigned laddr = ldsoff + boff;
        const void* gaddr = src + boff;
        asm volatile("global_load_async_to_lds_b128 %0, %1, off"
                     :: "v"(laddr), "v"(gaddr) : "memory");
      }
    }
    asm volatile("s_wait_asynccnt 0x0" ::: "memory");
  }
  __syncthreads();

  int wave = blockIdx.x * (blockDim.x >> 5) + (t >> 5);
  if (wave >= NT32) return;            // whole-wave uniform exit (after barrier)
  int b = blockIdx.y;                  // batch (no integer division)
  int node_base = wave * MT;

  int g  = lane >> 4;                  // lane group (selects K sub-range)
  int m  = lane & 15;                  // A-matrix row / C-matrix column
  int d0 = g * 8;                      // A chunk base: K in {8g..8g+7, 8g+16..8g+23}

  int row0 = node_base + m;                       // subtile0 row (always valid)
  int row1 = node_base + 16 + m;                  // subtile1 row
  bool st1_valid = (node_base + 16) < NN;         // wave-uniform
  if (row1 >= NN) row1 = NN - 1;                  // clamp gathers for tail tile

  v8f acc00 = {};   // subtile0, F = 0..15
  v8f acc01 = {};   // subtile0, F = 16..31
  v8f acc10 = {};   // subtile1, F = 0..15
  v8f acc11 = {};   // subtile1, F = 16..31

  size_t xbase = (size_t)b * NN * DD;

#pragma unroll 4
  for (int k = 0; k < KNBR; ++k) {
    // neighbor row ids (int64 low dword; N < 2^31)
    int i0 = nbr2[2 * (row0 * KNBR + k)];
    int i1 = nbr2[2 * (row1 * KNBR + k)];
    size_t rb0 = xbase + (size_t)i0 * DD;
    size_t rb1 = xbase + (size_t)i1 * DD;

    v8bf a0h0 = *(const v8bf*)(xhi + rb0 + d0);
    v8bf a0h1 = *(const v8bf*)(xhi + rb0 + d0 + 16);
    v8bf a0l0 = *(const v8bf*)(xlo + rb0 + d0);
    v8bf a0l1 = *(const v8bf*)(xlo + rb0 + d0 + 16);
    v8bf a1h0 = *(const v8bf*)(xhi + rb1 + d0);
    v8bf a1h1 = *(const v8bf*)(xhi + rb1 + d0 + 16);
    v8bf a1l0 = *(const v8bf*)(xlo + rb1 + d0);
    v8bf a1l1 = *(const v8bf*)(xlo + rb1 + d0 + 16);
    v16bf a0_hi = __builtin_shufflevector(a0h0, a0h1, 0,1,2,3,4,5,6,7,8,9,10,11,12,13,14,15);
    v16bf a0_lo = __builtin_shufflevector(a0l0, a0l1, 0,1,2,3,4,5,6,7,8,9,10,11,12,13,14,15);
    v16bf a1_hi = __builtin_shufflevector(a1h0, a1h1, 0,1,2,3,4,5,6,7,8,9,10,11,12,13,14,15);
    v16bf a1_lo = __builtin_shufflevector(a1l0, a1l1, 0,1,2,3,4,5,6,7,8,9,10,11,12,13,14,15);

    // B operands from LDS (ds_load path; frees VMEM pipe for the gathers)
    int wb = k * 1024 + lane * 16;               // [k][fh][lane][16]
    v16bf bh0 = *(const v16bf*)(sw + wb);
    v16bf bh1 = *(const v16bf*)(sw + wb + 512);
    v16bf bl0 = *(const v16bf*)(sw + WELEM + wb);
    v16bf bl1 = *(const v16bf*)(sw + WELEM + wb + 512);

    // out += A_hi*B_hi + A_hi*B_lo + A_lo*B_hi  (split-precision bf16)
    acc00 = __builtin_amdgcn_wmma_f32_16x16x32_bf16(false, a0_hi, false, bh0, (short)0, acc00, false, false);
    acc00 = __builtin_amdgcn_wmma_f32_16x16x32_bf16(false, a0_hi, false, bl0, (short)0, acc00, false, false);
    acc00 = __builtin_amdgcn_wmma_f32_16x16x32_bf16(false, a0_lo, false, bh0, (short)0, acc00, false, false);
    acc01 = __builtin_amdgcn_wmma_f32_16x16x32_bf16(false, a0_hi, false, bh1, (short)0, acc01, false, false);
    acc01 = __builtin_amdgcn_wmma_f32_16x16x32_bf16(false, a0_hi, false, bl1, (short)0, acc01, false, false);
    acc01 = __builtin_amdgcn_wmma_f32_16x16x32_bf16(false, a0_lo, false, bh1, (short)0, acc01, false, false);
    acc10 = __builtin_amdgcn_wmma_f32_16x16x32_bf16(false, a1_hi, false, bh0, (short)0, acc10, false, false);
    acc10 = __builtin_amdgcn_wmma_f32_16x16x32_bf16(false, a1_hi, false, bl0, (short)0, acc10, false, false);
    acc10 = __builtin_amdgcn_wmma_f32_16x16x32_bf16(false, a1_lo, false, bh0, (short)0, acc10, false, false);
    acc11 = __builtin_amdgcn_wmma_f32_16x16x32_bf16(false, a1_hi, false, bh1, (short)0, acc11, false, false);
    acc11 = __builtin_amdgcn_wmma_f32_16x16x32_bf16(false, a1_hi, false, bl1, (short)0, acc11, false, false);
    acc11 = __builtin_amdgcn_wmma_f32_16x16x32_bf16(false, a1_lo, false, bh1, (short)0, acc11, false, false);
  }

  // Epilogue: C/D layout -> lane = column, VGPR r = row (r + 8g)
  float bz0 = bias[m];
  float bz1 = bias[m + 16];
  size_t ob0 = ((size_t)b * NN + (size_t)node_base) * FF;
#pragma unroll
  for (int r = 0; r < 8; ++r) {
    int M = r + 8 * g;
    out[ob0 + (size_t)M * FF + m]      = fmaxf(acc00[r] + bz0, 0.0f);
    out[ob0 + (size_t)M * FF + 16 + m] = fmaxf(acc01[r] + bz1, 0.0f);
  }
  if (st1_valid) {
    size_t ob1 = ob0 + (size_t)16 * FF;
#pragma unroll
    for (int r = 0; r < 8; ++r) {
      int M = r + 8 * g;
      out[ob1 + (size_t)M * FF + m]      = fmaxf(acc10[r] + bz0, 0.0f);
      out[ob1 + (size_t)M * FF + 16 + m] = fmaxf(acc11[r] + bz1, 0.0f);
    }
  }
}

extern "C" void kernel_launch(void* const* d_in, const int* in_sizes, int n_in,
                              void* d_out, int out_size, void* d_ws, size_t ws_size,
                              hipStream_t stream) {
  const float* x    = (const float*)d_in[0];           // [B][N][D] fp32
  const int*   nbr2 = (const int*)d_in[1];             // [N][K] int64, viewed as int pairs
  const float* w    = (const float*)d_in[2];           // [K][D][F] fp32
  const float* bias = (const float*)d_in[3];           // [F] fp32
  float*       out  = (float*)d_out;                   // [B][N][F] fp32

  const int xelems = BB * NN * DD;                     // 6,400,000
  unsigned short* xhi = (unsigned short*)d_ws;
  unsigned short* xlo = xhi + xelems;
  unsigned short* whi = xlo + xelems;
  unsigned short* wlo = whi + WELEM;

  cvt_x_kernel<<<(xelems + 255) / 256, 256, 0, stream>>>(x, xhi, xlo, xelems);
  pack_w_kernel<<<(WELEM + 255) / 256, 256, 0, stream>>>(w, whi, wlo);

  dim3 grid((NT32 + 7) / 8, BB);                       // 8 waves (256 thr) per block
  gconv_wmma_kernel<<<grid, 256, 0, stream>>>(xhi, xlo, whi, wlo, nbr2, bias, out);
}
